// GalaxyTokenEncoder_47150150975902
// MI455X (gfx1250) — matrix operand order repaired and checked by reading, not compiled, not run on hardware
//
#include <hip/hip_runtime.h>

typedef __attribute__((ext_vector_type(16))) _Float16 v16h;
typedef __attribute__((ext_vector_type(8)))  _Float16 v8h;
typedef __attribute__((ext_vector_type(4)))  _Float16 h4;
typedef __attribute__((ext_vector_type(8)))  float    v8f;
typedef __attribute__((ext_vector_type(4)))  float    f4;
typedef __attribute__((ext_vector_type(4)))  unsigned int u32x4;
typedef __attribute__((ext_vector_type(8)))  int          i32x8;
typedef __attribute__((ext_vector_type(4)))  int          i32x4;

#define NTOK     64          // tokens per block (M tile)
#define XSTRIDE  520         // halves per X row (512 + 8 pad -> 4-bank rotate/row)
#define KPAD1    512         // layer-1 K padded (454 -> 512 with permuted layout)
#define NDIM     256         // output channels per layer
#define KCHUNK   64          // K chunk staged per step
#define WSTRIDE  72          // halves per W row in LDS (64 data + 8 pad)

#define X_BYTES    (NTOK * XSTRIDE * 2)
#define W_BYTES    (NDIM * WSTRIDE * 2)
#define LDSOFF_W0  (X_BYTES)
#define LDSOFF_W1  (X_BYTES + W_BYTES)
#define LDSOFF_MSK (X_BYTES + 2 * W_BYTES)
#define SMEM_BYTES (X_BYTES + 2 * W_BYTES + NTOK * 4)

// prepped weight sizes (halves) in d_ws: WT layout [n][k], f16
#define S_WT1 (NDIM * KPAD1)
#define S_WT2 (NDIM * NDIM)

#if defined(__HIP_DEVICE_COMPILE__) && __has_builtin(__builtin_amdgcn_tensor_load_to_lds)
#define HAS_TDM 1
#else
#define HAS_TDM 0
#endif

static __device__ __forceinline__ int clampi(int v, int lo, int hi) {
  return v < lo ? lo : (v > hi ? hi : v);
}

static __device__ __forceinline__ float fast_tanh(float x) {
#if defined(__HIP_DEVICE_COMPILE__) && __has_builtin(__builtin_amdgcn_tanhf)
  return __builtin_amdgcn_tanhf(x);
#else
  return tanhf(x);
#endif
}

static __device__ __forceinline__ float geluf(float x) {
  float x3 = x * x * x;
  return 0.5f * x * (1.0f + fast_tanh(0.7978845608028654f * (x + 0.044715f * x3)));
}

static __device__ __forceinline__ void wait_tensorcnt0() {
#if defined(__HIP_DEVICE_COMPILE__) && __has_builtin(__builtin_amdgcn_s_wait_tensorcnt)
  __builtin_amdgcn_s_wait_tensorcnt((short)0);
#elif defined(__HIP_DEVICE_COMPILE__)
  asm volatile("s_wait_tensorcnt 0x0" ::: "memory");
#endif
}

// A fragment: 16x32 f16, lane L -> row M = m_strip + (L&15),
// K = k + {0..7,16..23} (lanes<16) or k + {8..15,24..31} (lanes>=16)
static __device__ __forceinline__ v16h frag_a(const _Float16* xb, int m_strip, int k, int lane) {
  const _Float16* q = xb + (m_strip + (lane & 15)) * XSTRIDE + k + ((lane >> 4) << 3);
  v8h lo = *(const v8h*)q;
  v8h hi = *(const v8h*)(q + 16);
  return __builtin_shufflevector(lo, hi, 0,1,2,3,4,5,6,7,8,9,10,11,12,13,14,15);
}

// B fragment: 32x16 f16 from W^T tile in LDS ([N][KCHUNK] pad WSTRIDE):
// lane L -> col N = n + (L&15), K = kk + {0..7,16..23} (lanes<16) or kk + {8..15,24..31} (lanes>=16)
static __device__ __forceinline__ v16h frag_b(const _Float16* wb, int n, int kk, int lane) {
  const _Float16* q = wb + (n + (lane & 15)) * WSTRIDE + kk + ((lane >> 4) << 3);
  v8h lo = *(const v8h*)q;
  v8h hi = *(const v8h*)(q + 16);
  return __builtin_shufflevector(lo, hi, 0,1,2,3,4,5,6,7,8,9,10,11,12,13,14,15);
}

#if HAS_TDM
// One TDM descriptor: 2D tile, 256 rows x 128B (16 x 8B elements), row stride Kpad*2 B,
// into LDS at lds_off with 16B pad inserted after every 128B (-> WSTRIDE rows).
static __device__ __forceinline__ void issue_tdm(const _Float16* WT, int Kpad, int k0,
                                                 unsigned lds_off) {
  unsigned long long ga = (unsigned long long)(uintptr_t)(WT + k0);
  u32x4 g0;
  g0.x = 1u;                                               // count=1 (valid user D#)
  g0.y = lds_off;                                          // lds_addr (bytes)
  g0.z = (unsigned)ga;                                     // global_addr[31:0]
  g0.w = (unsigned)((ga >> 32) & 0x1FFFFFFull) | (2u << 30); // global_addr[56:32] | type=2
  i32x8 g1;
  g1[0] = (3 << 16)      // data_size = 8B
        | (1 << 20)      // pad_enable
        | (4 << 22)      // pad_interval: 32 DWORDs (=128B) between pads
        | (3 << 25);     // pad_amount: 4 DWORDs (=16B)
  g1[1] = (int)(16u << 16);   // tensor_dim0[15:0]=16 (8B units per row)
  g1[2] = (int)(256u << 16);  // tensor_dim0 hi=0 | tensor_dim1[15:0]=256 rows
  g1[3] = (int)(16u << 16);   // tensor_dim1 hi=0 | tile_dim0=16
  g1[4] = 256;                // tile_dim1=256 | tile_dim2=0
  g1[5] = Kpad / 4;           // tensor_dim0_stride low32 (8B units: Kpad*2/8)
  g1[6] = 0;
  g1[7] = 0;
  i32x4 z4 = {0, 0, 0, 0};
#if __clang_major__ >= 23
  i32x8 z8 = {0, 0, 0, 0, 0, 0, 0, 0};
  __builtin_amdgcn_tensor_load_to_lds(g0, g1, z4, z4, z8, 0);
#else
  __builtin_amdgcn_tensor_load_to_lds(g0, g1, z4, z4, 0);
#endif
}
#endif

static __device__ __forceinline__ void compute_chunk(v8f acc[8], const _Float16* xb,
                                                     const _Float16* wb, int k0,
                                                     int m_strip, int n_base, int lane) {
#pragma unroll
  for (int kk = 0; kk < KCHUNK; kk += 32) {
    // Batch all 18 ds_load_b128 before any WMMA so the scheduler can use
    // graduated s_wait_dscnt counts instead of wait-0 chains.
    v16h a = frag_a(xb, m_strip, k0 + kk, lane);
    v16h bf[8];
#pragma unroll
    for (int j = 0; j < 8; ++j) bf[j] = frag_b(wb, n_base + j * 16, kk, lane);
#pragma unroll
    for (int j = 0; j < 8; ++j)
      acc[j] = __builtin_amdgcn_wmma_f32_16x16x32_f16(
          false, a, false, bf[j], (short)0, acc[j], false, false);
  }
}

static __device__ __forceinline__ void layer_mm(v8f acc[8], const _Float16* xb,
                                                _Float16* wb0, _Float16* wb1,
                                                const _Float16* __restrict__ WT, int Kpad,
                                                int m_strip, int n_base, int lane,
                                                int wid, int tid) {
  const int nchunk = Kpad / KCHUNK;
#if HAS_TDM
  if (wid == 0) issue_tdm(WT, Kpad, 0, LDSOFF_W0);         // prologue fill buf0
  for (int c = 0; c < nchunk; ++c) {
    _Float16* wb = (c & 1) ? wb1 : wb0;
    if (wid == 0) wait_tensorcnt0();                       // chunk c resident
    __syncthreads();                                       // visible to all waves
    if (wid == 0 && (c + 1) < nchunk)                      // overlap next DMA w/ compute
      issue_tdm(WT, Kpad, (c + 1) * KCHUNK, ((c + 1) & 1) ? LDSOFF_W1 : LDSOFF_W0);
    compute_chunk(acc, xb, wb, c * KCHUNK, m_strip, n_base, lane);
  }
#else
  for (int c = 0; c < nchunk; ++c) {
    int k0 = c * KCHUNK;
    __syncthreads();
    for (int e = tid; e < NDIM * (KCHUNK / 8); e += 256) { // 2048 b128 copies
      int n = e >> 3, g = e & 7;
      *(v8h*)(wb0 + n * WSTRIDE + g * 8) =
          *(const v8h*)(WT + (size_t)n * Kpad + k0 + g * 8);
    }
    __syncthreads();
    compute_chunk(acc, xb, wb0, k0, m_strip, n_base, lane);
  }
#endif
  __syncthreads();  // all waves done reading X/W before epilogue overwrites X
}

// ---------------- prep: transpose + f16-convert weights into workspace ----------------
// WT1: [256][512] f16, K permuted: 0..3 gal, 4..5 pix, 6..7 zero, 8..71 feat0,
//      72..199 feat1, 200..455 feat2, 456..511 zero
// WT2/WT3: [256][256] f16
__global__ __launch_bounds__(256) void prep_weights(const float* __restrict__ W1,
                                                    const float* __restrict__ W2,
                                                    const float* __restrict__ W3,
                                                    _Float16* __restrict__ wt) {
  int idx = blockIdx.x * 256 + threadIdx.x;
  if (idx < S_WT1) {
    int n = idx >> 9, k = idx & 511;
    float v = 0.0f;
    if (k < 6)                   v = W1[k * NDIM + n];
    else if (k >= 8 && k < 456)  v = W1[(k - 2) * NDIM + n];
    wt[idx] = (_Float16)v;
  } else if (idx < S_WT1 + S_WT2) {
    int e = idx - S_WT1;
    int n = e >> 8, k = e & 255;
    wt[idx] = (_Float16)W2[k * NDIM + n];
  } else if (idx < S_WT1 + 2 * S_WT2) {
    int e = idx - S_WT1 - S_WT2;
    int n = e >> 8, k = e & 255;
    wt[idx] = (_Float16)W3[k * NDIM + n];
  }
}

// ---------------- fused gather + 3-layer MLP ----------------
__global__ __launch_bounds__(256) void galaxy_token_encoder(
    const float* __restrict__ gal, const float* __restrict__ coords,
    const float* __restrict__ f0, const float* __restrict__ f1,
    const float* __restrict__ f2, const float* __restrict__ msk,
    const _Float16* __restrict__ wt,
    const float* __restrict__ b1, const float* __restrict__ b2,
    const float* __restrict__ b3,
    const int* __restrict__ cisp, float* __restrict__ out) {
  extern __shared__ unsigned char smem[];
  _Float16* lds_x   = (_Float16*)smem;                 // 64 x 520 halves
  _Float16* lds_w0  = (_Float16*)(smem + LDSOFF_W0);   // 256 x 72 halves (buf 0)
  _Float16* lds_w1  = (_Float16*)(smem + LDSOFF_W1);   // 256 x 72 halves (buf 1)
  float*    lds_msk = (float*)(smem + LDSOFF_MSK);     // 64 floats

  const int tid  = threadIdx.x;
  const int lane = tid & 31;
  const int wid  = tid >> 5;
  const int m_strip = (wid & 3) * 16;
  const int n_base  = (wid >> 2) * 128;
  const int t0 = blockIdx.x * NTOK;

  const int   cis   = cisp[0];
  const float denom = (float)((cis - 1) > 1 ? (cis - 1) : 1);
  const float inv   = 1.0f / denom;

  const _Float16* WT1 = wt;
  const _Float16* WT2 = wt + S_WT1;
  const _Float16* WT3 = wt + S_WT1 + S_WT2;

  // ---------------- gather: build X tile (f16, permuted K) ----------------
  {
    const int i = tid >> 2;            // token within tile; 4 threads per token
    const int j = tid & 3;
    const int t = t0 + i;
    const int b = t >> 13;             // N = 8192
    const float px = coords[t * 2 + 0];
    const float py = coords[t * 2 + 1];
    _Float16* row = lds_x + i * XSTRIDE;

    if (j == 0) {
      row[0] = (_Float16)gal[t * 4 + 0];
      row[1] = (_Float16)gal[t * 4 + 1];
      row[2] = (_Float16)gal[t * 4 + 2];
      row[3] = (_Float16)gal[t * 4 + 3];
      row[4] = (_Float16)(2.0f * px * inv - 1.0f);
      row[5] = (_Float16)(2.0f * py * inv - 1.0f);
      row[6] = (_Float16)0.0f;
      row[7] = (_Float16)0.0f;
    }
    if (j == 1) lds_msk[i] = msk[t];

    h4 zz = {};
    for (int c4 = 114 + j; c4 < 128; c4 += 4)            // zero 456..511
      *(h4*)(row + c4 * 4) = zz;

    {  // feat0: (B,128,128,64) -> cols 8..71
      int xi = clampi((int)rintf(px * (127.0f * inv)), 0, 127);
      int yi = clampi((int)rintf(py * (127.0f * inv)), 0, 127);
      const f4* p = (const f4*)(f0 + (size_t)(((b * 128) + yi) * 128 + xi) * 64);
      for (int c = j; c < 16; c += 4) {
        f4 v = p[c];
        h4 h; h.x = (_Float16)v.x; h.y = (_Float16)v.y; h.z = (_Float16)v.z; h.w = (_Float16)v.w;
        *(h4*)(row + 8 + c * 4) = h;
      }
    }
    {  // feat1: (B,64,64,128) -> cols 72..199
      int xi = clampi((int)rintf(px * (63.0f * inv)), 0, 63);
      int yi = clampi((int)rintf(py * (63.0f * inv)), 0, 63);
      const f4* p = (const f4*)(f1 + (size_t)(((b * 64) + yi) * 64 + xi) * 128);
      for (int c = j; c < 32; c += 4) {
        f4 v = p[c];
        h4 h; h.x = (_Float16)v.x; h.y = (_Float16)v.y; h.z = (_Float16)v.z; h.w = (_Float16)v.w;
        *(h4*)(row + 72 + c * 4) = h;
      }
    }
    {  // feat2: (B,32,32,256) -> cols 200..455
      int xi = clampi((int)rintf(px * (31.0f * inv)), 0, 31);
      int yi = clampi((int)rintf(py * (31.0f * inv)), 0, 31);
      const f4* p = (const f4*)(f2 + (size_t)(((b * 32) + yi) * 32 + xi) * 256);
      for (int c = j; c < 64; c += 4) {
        f4 v = p[c];
        h4 h; h.x = (_Float16)v.x; h.y = (_Float16)v.y; h.z = (_Float16)v.z; h.w = (_Float16)v.w;
        *(h4*)(row + 200 + c * 4) = h;
      }
    }
  }
  __syncthreads();

  v8f acc[8];
  const int ncol_lo = lane & 15;

  // ---------------- layer 1 ----------------
#pragma unroll
  for (int j = 0; j < 8; ++j) {
    float bv = b1[n_base + j * 16 + ncol_lo];
    v8f z; for (int e = 0; e < 8; ++e) z[e] = bv;
    acc[j] = z;
  }
  layer_mm(acc, lds_x, lds_w0, lds_w1, WT1, KPAD1, m_strip, n_base, lane, wid, tid);
#pragma unroll
  for (int j = 0; j < 8; ++j) {
    int n = n_base + j * 16 + ncol_lo;
#pragma unroll
    for (int v = 0; v < 8; ++v) {
      int m = m_strip + v + ((lane >> 4) << 3);
      lds_x[m * XSTRIDE + n] = (_Float16)geluf(acc[j][v]);
    }
  }
  __syncthreads();

  // ---------------- layer 2 ----------------
#pragma unroll
  for (int j = 0; j < 8; ++j) {
    float bv = b2[n_base + j * 16 + ncol_lo];
    v8f z; for (int e = 0; e < 8; ++e) z[e] = bv;
    acc[j] = z;
  }
  layer_mm(acc, lds_x, lds_w0, lds_w1, WT2, NDIM, m_strip, n_base, lane, wid, tid);
#pragma unroll
  for (int j = 0; j < 8; ++j) {
    int n = n_base + j * 16 + ncol_lo;
#pragma unroll
    for (int v = 0; v < 8; ++v) {
      int m = m_strip + v + ((lane >> 4) << 3);
      lds_x[m * XSTRIDE + n] = (_Float16)geluf(acc[j][v]);
    }
  }
  __syncthreads();

  // ---------------- layer 3 + mask + store ----------------
#pragma unroll
  for (int j = 0; j < 8; ++j) {
    float bv = b3[n_base + j * 16 + ncol_lo];
    v8f z; for (int e = 0; e < 8; ++e) z[e] = bv;
    acc[j] = z;
  }
  layer_mm(acc, lds_x, lds_w0, lds_w1, WT3, NDIM, m_strip, n_base, lane, wid, tid);
#pragma unroll
  for (int j = 0; j < 8; ++j) {
    int n = n_base + j * 16 + ncol_lo;
#pragma unroll
    for (int v = 0; v < 8; ++v) {
      int m = m_strip + v + ((lane >> 4) << 3);
      float g = geluf(acc[j][v]) * lds_msk[m];
      out[(size_t)(t0 + m) * NDIM + n] = g;
    }
  }
}

extern "C" void kernel_launch(void* const* d_in, const int* in_sizes, int n_in,
                              void* d_out, int out_size, void* d_ws, size_t ws_size,
                              hipStream_t stream) {
  const float* gal    = (const float*)d_in[0];
  const float* coords = (const float*)d_in[1];
  const float* f0     = (const float*)d_in[2];
  const float* f1     = (const float*)d_in[3];
  const float* f2     = (const float*)d_in[4];
  const float* msk    = (const float*)d_in[5];
  const float* W1     = (const float*)d_in[6];
  const float* b1     = (const float*)d_in[7];
  const float* W2     = (const float*)d_in[8];
  const float* b2     = (const float*)d_in[9];
  const float* W3     = (const float*)d_in[10];
  const float* b3     = (const float*)d_in[11];
  const int*   cis    = (const int*)d_in[12];
  float* out = (float*)d_out;
  _Float16* wt = (_Float16*)d_ws;   // needs (512+256+256)*256*2 = 512 KB of scratch

  const int prep_elems = S_WT1 + 2 * S_WT2;               // 262144 halves
  prep_weights<<<prep_elems / 256, 256, 0, stream>>>(W1, W2, W3, wt);

  const int total_tokens = in_sizes[5];                   // B*N = 65536
  const int blocks = (total_tokens + NTOK - 1) / NTOK;
  galaxy_token_encoder<<<blocks, 256, SMEM_BYTES, stream>>>(
      gal, coords, f0, f1, f2, msk, wt, b1, b2, b3, cis, out);
}
